// PiSSAAdapter_52836687675854
// MI455X (gfx1250) — matrix operand order repaired and checked by reading, not compiled, not run on hardware
//
#include <hip/hip_runtime.h>

// ---------------- problem dims (fixed by the reference) ----------------
#define Mdim 16384   // B*S = 8*2048
#define Kdim 4096    // IN
#define Ndim 4096    // OUT
#define Rdim 64      // LoRA rank

typedef __attribute__((ext_vector_type(16))) __bf16 v16bf;
typedef __attribute__((ext_vector_type(8)))  __bf16 v8bf;
typedef __attribute__((ext_vector_type(4)))  __bf16 v4bf;
typedef __attribute__((ext_vector_type(8)))  float  v8f;
typedef int v4i_g __attribute__((vector_size(16)));        // builtin V4i
typedef unsigned int tdm_v4u __attribute__((vector_size(16)));
typedef int tdm_v8i __attribute__((vector_size(32)));
typedef int tdm_v4i __attribute__((vector_size(16)));

// ---------------- gfx1250 feature probes ----------------
#if defined(__HIP_DEVICE_COMPILE__) && __has_builtin(__builtin_amdgcn_global_load_async_to_lds_b128)
#define HAVE_ASYNC_LDS 1
#else
#define HAVE_ASYNC_LDS 0
#endif

#if defined(__HIP_DEVICE_COMPILE__) && __has_builtin(__builtin_amdgcn_tensor_load_to_lds)
#define HAVE_TDM 1
#else
#define HAVE_TDM 0
#endif

__device__ __forceinline__ void async_copy_b128(const void* g, void* l) {
#if HAVE_ASYNC_LDS
    __builtin_amdgcn_global_load_async_to_lds_b128(
        (__attribute__((address_space(1))) v4i_g*)(void*)g,
        (__attribute__((address_space(3))) v4i_g*)l,
        /*imm offset*/0, /*cpol*/0);
#else
    *(v8bf*)l = *(const v8bf*)g;   // sync fallback: global load + ds_store
#endif
}

__device__ __forceinline__ void wait_async_zero() {
#if HAVE_ASYNC_LDS
#if __has_builtin(__builtin_amdgcn_s_wait_asynccnt)
    __builtin_amdgcn_s_wait_asynccnt(0);
#else
    asm volatile("s_wait_asynccnt 0x0" ::: "memory");
#endif
#endif
}

__device__ __forceinline__ void wait_tensor_zero() {
#if HAVE_TDM
#if __has_builtin(__builtin_amdgcn_s_wait_tensorcnt)
    __builtin_amdgcn_s_wait_tensorcnt(0);
#else
    asm volatile("s_wait_tensorcnt 0x0" ::: "memory");
#endif
#endif
}

// =====================================================================
// Kernel 1: W_eff[o,i] = W_res[o,i] + sum_r lora_B[o,r] * lora_A[r,i]
// stored as bf16 in workspace. ~2.1 GFLOP, lora_A/B cache resident.
// =====================================================================
__global__ __launch_bounds__(256) void build_weff_kernel(
    const float* __restrict__ Wres,
    const float* __restrict__ lA,    // [R, K]
    const float* __restrict__ lB,    // [N, R]
    __bf16* __restrict__ Weff)       // [N, K] bf16
{
    const int tid = blockIdx.x * blockDim.x + threadIdx.x;
    const int i = (tid % (Kdim / 4)) * 4;
    const int o = (tid / (Kdim / 4)) * 2;

    float4 w0 = *(const float4*)(Wres + (size_t)o * Kdim + i);
    float4 w1 = *(const float4*)(Wres + (size_t)(o + 1) * Kdim + i);
    float acc0[4] = {w0.x, w0.y, w0.z, w0.w};
    float acc1[4] = {w1.x, w1.y, w1.z, w1.w};

    #pragma unroll 8
    for (int r = 0; r < Rdim; ++r) {
        const float4 a = *(const float4*)(lA + (size_t)r * Kdim + i);
        const float b0 = lB[(size_t)o * Rdim + r];
        const float b1 = lB[(size_t)(o + 1) * Rdim + r];
        acc0[0] += b0 * a.x; acc0[1] += b0 * a.y; acc0[2] += b0 * a.z; acc0[3] += b0 * a.w;
        acc1[0] += b1 * a.x; acc1[1] += b1 * a.y; acc1[2] += b1 * a.z; acc1[3] += b1 * a.w;
    }
    v4bf p0 = {(__bf16)acc0[0], (__bf16)acc0[1], (__bf16)acc0[2], (__bf16)acc0[3]};
    v4bf p1 = {(__bf16)acc1[0], (__bf16)acc1[1], (__bf16)acc1[2], (__bf16)acc1[3]};
    *(v4bf*)(Weff + (size_t)o * Kdim + i)       = p0;
    *(v4bf*)(Weff + (size_t)(o + 1) * Kdim + i) = p1;
}

// =====================================================================
// Kernel 2: Out[m,n] = sum_k x[m,k]*Weff[n,k] + bias[n]
// Block tile 128(M) x 256(N), BK=32, 8 wave32s in a 2(M)x4(N) grid;
// each wave owns 64x64 = 4x4 v_wmma_f32_16x16x32_bf16 accumulators.
// Double-buffered LDS:
//   - B tile (bf16 weights) DMA'd global->LDS by the Tensor Data Mover
//     (one tensor_load_to_lds per 128x32 half-tile, TDM pad_enable
//     reproducing the 80B padded row stride; TENSORcnt tracked),
//     falling back to GLOBAL_LOAD_ASYNC_TO_LDS / sync copies.
//   - A tile software-pipelined through VGPRs (load t+1, compute t,
//     cvt f32->bf16 + ds_store t+1), one wait + barrier per iteration.
// =====================================================================
#define BM 128
#define BN 256
#define BK 32
#define LDA 40   // padded LDS row stride (bf16 elems): 80B rows, 16B aligned
#define LDB 40
#define KSTEPS (Kdim / BK)   // 128

__global__ __launch_bounds__(256) void lora_gemm_kernel(
    const float*  __restrict__ X,     // [M, K] f32
    const __bf16* __restrict__ Weff,  // [N, K] bf16
    const float*  __restrict__ bias,  // [N]
    float*        __restrict__ Out)   // [M, N]
{
    __shared__ __align__(16) __bf16 sA[2][BM * LDA];   // 2 x 10 KB
    __shared__ __align__(16) __bf16 sB[2][BN * LDB];   // 2 x 20 KB

    const int t    = threadIdx.x;
    const int lane = t & 31;
    const int wave = t >> 5;
    const int wm   = wave >> 2;  // 0..1 -> M offset wm*64
    const int wn   = wave & 3;   // 0..3 -> N offset wn*64
    const int m0   = blockIdx.y * BM;
    const int n0   = blockIdx.x * BN;

    // staging decomposition
    const int arow   = t >> 3;   // 0..31  x: 8 float4 chunks / 32-col row
    const int achunk = t & 7;
    const int brow   = t >> 2;   // 0..63  (async fallback only)
    const int bchunk = t & 3;

    // WMMA lane decomposition (ISA 16-bit A/B layouts)
    const int hi = lane >> 4;
    const int lr = lane & 15;

    v8f acc[4][4] = {};
    float4 areg[4];

    const float*  aptr = X    + (size_t)(m0 + arow) * Kdim + achunk * 4;
    const __bf16* bptr = Weff + (size_t)(n0 + brow) * Kdim + bchunk * 8;
    (void)bptr;

    auto gload_A = [&](int k0) {
        #pragma unroll
        for (int p = 0; p < 4; ++p)
            areg[p] = *(const float4*)(aptr + (size_t)p * 32 * Kdim + k0);
    };
    auto store_A = [&](int b) {
        #pragma unroll
        for (int p = 0; p < 4; ++p) {
            v4bf pv = {(__bf16)areg[p].x, (__bf16)areg[p].y,
                       (__bf16)areg[p].z, (__bf16)areg[p].w};
            *(v4bf*)&sA[b][(arow + p * 32) * LDA + achunk * 4] = pv;
        }
    };

    auto stage_B = [&](int b, int k0) {
#if HAVE_TDM
        if ((wave & 3) == 0) {                 // waves 0 and 4: one half each
            const int h = wave >> 2;           // 0 or 1
            const __bf16* gt = Weff + (size_t)(n0 + h * 128) * Kdim + k0;
            const uint64_t ga = (uint64_t)(uintptr_t)gt;
            const uint32_t lds_addr = (uint32_t)(size_t)
                (__attribute__((address_space(3))) char*)&sB[b][h * 128 * LDB];
            // ---- D# group 0: count=1 | lds_addr | global_addr | type=2 ----
            tdm_v4u g0;
            g0[0] = 1u;
            g0[1] = lds_addr;
            g0[2] = (uint32_t)ga;
            g0[3] = (uint32_t)(ga >> 32) | (2u << 30);
            // ---- D# group 1 ----
            // data_size=1 (2B) | pad_enable | pad_interval=3 (16 DW = 64B row)
            // | pad_amount=3 (4 DW = 16B) -> 80B LDS rows matching LDB=40
            tdm_v8i g1;
            g1[0] = (1 << 16) | (1 << 20) | (3 << 22) | (3 << 25);
            g1[1] = (int)(4096u << 16);   // tensor_dim0 = 4096 (bits 79:48)
            g1[2] = (int)0x10000000;      // tensor_dim1 = 4096 (bits 111:80)
            g1[3] = 32 << 16;             // tile_dim0 = 32 (bits 127:112)
            g1[4] = 128;                  // tile_dim1 = 128 rows
            g1[5] = 4096;                 // tensor_dim0_stride = 4096 elems
            g1[6] = 0;
            g1[7] = 0;
            tdm_v4i z4 = {0, 0, 0, 0};            // groups 2/3 unused (2D tile)
            tdm_v8i z8 = {0, 0, 0, 0, 0, 0, 0, 0};
            __builtin_amdgcn_tensor_load_to_lds(g0, g1, z4, z4, z8, /*cpol*/0);
        }
#else
        #pragma unroll
        for (int p = 0; p < 4; ++p)
            async_copy_b128(bptr + (size_t)p * 64 * Kdim + k0,
                            &sB[b][(brow + p * 64) * LDB + bchunk * 8]);
#endif
    };

    auto wait_staged_B = [&]() {
#if HAVE_TDM
        wait_tensor_zero();
#else
        wait_async_zero();
#endif
    };

    // ---- prologue: fill buffer 0 ----
    gload_A(0);
    stage_B(0, 0);
    store_A(0);
    wait_staged_B();
    __syncthreads();

    int buf = 0;
    for (int step = 0; step < KSTEPS; ++step) {
        const bool more = (step + 1 < KSTEPS);
        if (more) {
            gload_A((step + 1) * BK);        // in flight during compute
            stage_B(buf ^ 1, (step + 1) * BK);
        }

        // ---- fragments from LDS[buf] ----
        // A 16x32: lane lr = row M; half 'hi': K = hi*8+[0..7] and 16+hi*8+[0..7]
        v16bf af[4];
        #pragma unroll
        for (int i2 = 0; i2 < 4; ++i2) {
            const __bf16* base = &sA[buf][(wm * 64 + i2 * 16 + lr) * LDA];
            const v8bf c0 = *(const v8bf*)(base + hi * 8);
            const v8bf c1 = *(const v8bf*)(base + 16 + hi * 8);
            #pragma unroll
            for (int e = 0; e < 8; ++e) { af[i2][e] = c0[e]; af[i2][e + 8] = c1[e]; }
        }
        // B 32x16: lane lr = column N; half 'hi': K = hi*8+[0..7] and 16+hi*8+[0..7]
        v16bf bfr[4];
        #pragma unroll
        for (int j = 0; j < 4; ++j) {
            const __bf16* base = &sB[buf][(wn * 64 + j * 16 + lr) * LDB + hi * 8];
            const v8bf c0 = *(const v8bf*)(base);
            const v8bf c1 = *(const v8bf*)(base + 16);
            #pragma unroll
            for (int e = 0; e < 8; ++e) { bfr[j][e] = c0[e]; bfr[j][e + 8] = c1[e]; }
        }

        // ---- 16 independent WMMAs ----
        #pragma unroll
        for (int i2 = 0; i2 < 4; ++i2)
            #pragma unroll
            for (int j = 0; j < 4; ++j)
                acc[i2][j] = __builtin_amdgcn_wmma_f32_16x16x32_bf16(
                    false, af[i2], false, bfr[j], (short)0, acc[i2][j],
                    false, false);

        if (more) store_A(buf ^ 1);   // cvt + ds_store after compute issued
        wait_staged_B();
        __syncthreads();
        buf ^= 1;
    }

    // ---- epilogue: C layout lane=col, VGPR=row (+8 for lanes 16-31) ----
    #pragma unroll
    for (int i2 = 0; i2 < 4; ++i2) {
        const int rowb = m0 + wm * 64 + i2 * 16 + hi * 8;
        #pragma unroll
        for (int j = 0; j < 4; ++j) {
            const int col = n0 + wn * 64 + j * 16 + lr;
            const float bv = bias[col];
            #pragma unroll
            for (int r = 0; r < 8; ++r)
                Out[(size_t)(rowb + r) * Ndim + col] = acc[i2][j][r] + bv;
        }
    }
}

// =====================================================================
extern "C" void kernel_launch(void* const* d_in, const int* in_sizes, int n_in,
                              void* d_out, int out_size, void* d_ws, size_t ws_size,
                              hipStream_t stream) {
    const float* x    = (const float*)d_in[0];  // [8,2048,4096]
    const float* Wres = (const float*)d_in[1];  // [4096,4096]
    const float* bias = (const float*)d_in[2];  // [4096]
    const float* lA   = (const float*)d_in[3];  // [64,4096]
    const float* lB   = (const float*)d_in[4];  // [4096,64]
    float* out  = (float*)d_out;                // [8,2048,4096]
    __bf16* Weff = (__bf16*)d_ws;               // 4096*4096*2 = 33.5 MB

    const int t1 = (Kdim / 4) * (Ndim / 2);
    build_weff_kernel<<<t1 / 256, 256, 0, stream>>>(Wres, lA, lB, Weff);

    dim3 grid(Ndim / BN, Mdim / BM);            // (16, 128) n-fastest
    lora_gemm_kernel<<<grid, 256, 0, stream>>>(x, Weff, bias, out);
}
